// MolGNet_72335839199972
// MI455X (gfx1250) — matrix operand order repaired and checked by reading, compiled, not run on hardware
//
#include <hip/hip_runtime.h>
#include <hip/hip_bf16.h>

// ---------------- problem constants (fixed by the reference) ----------------
#define D_IN 78
#define DD   768
#define LL   5
#define BN_EPS 1e-5f

// ---------------- WMMA types ----------------
typedef __attribute__((ext_vector_type(16))) __bf16 v16bf;
typedef __attribute__((ext_vector_type(8)))  float  v8f;

// fast GEMM tile
#define FBM 128
#define FBN 128
#define FBK 32
#define LDA 40   // bf16 elems per A row in LDS (32 + 8 pad; 16B-aligned rows)
#define LDB 136  // bf16 elems per B row in LDS (128 + 8 pad; 16B-aligned rows)

// ---------------- CDNA5 async global->LDS helpers ----------------
__device__ __forceinline__ unsigned lds_off(const void* p) {
    // generic pointer to LDS: low 32 bits are the DS byte offset
    return (unsigned)(size_t)p;
}
__device__ __forceinline__ void async_copy_b128(unsigned lds, const void* gptr) {
    asm volatile("global_load_async_to_lds_b128 %0, %1, off"
                 :: "v"(lds), "v"(gptr)
                 : "memory");
}
__device__ __forceinline__ void wait_async0() {
#if __has_builtin(__builtin_amdgcn_s_wait_asynccnt)
    __builtin_amdgcn_s_wait_asynccnt(0);
#else
    asm volatile("s_wait_asynccnt 0" ::: "memory");
#endif
}

// ============================================================================
// FAST GEMM: C[M,Nn] = A[M,K] x B[K,Nn], A/B bf16, f32 accumulate.
// Requires K%32==0, Nn%128==0 (all D=768 GEMMs). Double-buffered LDS filled by
// global_load_async_to_lds_b128 (ASYNCcnt), one barrier per K-step, 8
// back-to-back WMMAs per wave per step (wave tile 32Mx64N).
// Epilogue: C = acc (+bias[n]); optionally agg = C*dinv[m]^2 + cb[n].
// ============================================================================
__global__ __launch_bounds__(256)
void k_gemm_fast(const __bf16* __restrict__ A, const __bf16* __restrict__ B,
                 const float* __restrict__ bias, float* __restrict__ C,
                 float* __restrict__ agg, const float* __restrict__ dinv,
                 const float* __restrict__ cb, int M, int K, int Nn) {
    __shared__ __bf16 sA[2][FBM * LDA];
    __shared__ __bf16 sB[2][FBK * LDB];

    const int tid  = threadIdx.x;
    const int lane = tid & 31;
    const int wave = tid >> 5;
    const int wm   = wave & 3;     // 32-row group
    const int wn   = wave >> 2;    // 64-col group
    const int block_m = blockIdx.y * FBM;
    const int block_n = blockIdx.x * FBN;

    // staging coordinates (branchless; rows clamped to M-1)
    const int ar0 = tid >> 2;              // A rows 0..63 and 64..127
    const int ar1 = ar0 + 64;
    const int ac8 = (tid & 3) * 8;         // 8 bf16 = 16B chunk
    const int agm0 = min(block_m + ar0, M - 1);
    const int agm1 = min(block_m + ar1, M - 1);
    const int br0 = tid >> 4;              // B rows 0..15 and 16..31
    const int bc8 = (tid & 15) * 8;

    const __bf16* Abase0 = A + (size_t)agm0 * K + ac8;
    const __bf16* Abase1 = A + (size_t)agm1 * K + ac8;
    const __bf16* Bbase0 = B + (size_t)br0 * Nn + block_n + bc8;
    const __bf16* Bbase1 = B + (size_t)(br0 + 16) * Nn + block_n + bc8;

    // LDS destinations (per-buffer byte stride)
    const unsigned sa_o0 = lds_off(&sA[0][ar0 * LDA + ac8]);
    const unsigned sa_o1 = lds_off(&sA[0][ar1 * LDA + ac8]);
    const unsigned sb_o0 = lds_off(&sB[0][br0 * LDB + bc8]);
    const unsigned sb_o1 = lds_off(&sB[0][(br0 + 16) * LDB + bc8]);
    const unsigned sa_bs = FBM * LDA * sizeof(__bf16);
    const unsigned sb_bs = FBK * LDB * sizeof(__bf16);

    v8f zero = {0.f,0.f,0.f,0.f,0.f,0.f,0.f,0.f};
    v8f acc[2][4];
    #pragma unroll
    for (int g = 0; g < 2; ++g)
        #pragma unroll
        for (int t = 0; t < 4; ++t) acc[g][t] = zero;

    // prologue: async-stage ks=0 into buffer 0
    async_copy_b128(sa_o0, Abase0);
    async_copy_b128(sa_o1, Abase1);
    async_copy_b128(sb_o0, Bbase0);
    async_copy_b128(sb_o1, Bbase1);
    wait_async0();
    __syncthreads();

    const int ksteps = K / FBK;
    for (int ks = 0; ks < ksteps; ++ks) {
        const int cur = ks & 1, nxt = cur ^ 1;
        const int kn  = min(ks + 1, ksteps - 1) * FBK;

        // ---- kick off async copies for the next stage (background DMA) ----
        async_copy_b128(sa_o0 + nxt * sa_bs, Abase0 + kn);
        async_copy_b128(sa_o1 + nxt * sa_bs, Abase1 + kn);
        async_copy_b128(sb_o0 + nxt * sb_bs, Bbase0 + (size_t)kn * Nn);
        async_copy_b128(sb_o1 + nxt * sb_bs, Bbase1 + (size_t)kn * Nn);

        // ---- A fragments (16x32): half selects K runs {0..7,16..23}/{8..15,24..31}
        const int half = lane >> 4;
        const int am0  = wm * 32 + (lane & 15);
        v16bf af0, af1;
        #pragma unroll
        for (int j = 0; j < 8; ++j) {
            af0[j]     = sA[cur][am0 * LDA + half * 8 + j];
            af0[8 + j] = sA[cur][am0 * LDA + 16 + half * 8 + j];
            af1[j]     = sA[cur][(am0 + 16) * LDA + half * 8 + j];
            af1[8 + j] = sA[cur][(am0 + 16) * LDA + 16 + half * 8 + j];
        }
        // ---- preload all 4 B fragments (lane%16 = N col, K runs {0..7,16..23}/{8..15,24..31}) ----
        v16bf bfr[4];
        #pragma unroll
        for (int t = 0; t < 4; ++t) {
            const int nb = wn * 64 + t * 16;
            #pragma unroll
            for (int j = 0; j < 8; ++j) {
                bfr[t][j]     = sB[cur][(half * 8 + j) * LDB + nb + (lane & 15)];
                bfr[t][8 + j] = sB[cur][(16 + half * 8 + j) * LDB + nb + (lane & 15)];
            }
        }
        // ---- 8 back-to-back WMMAs ----
        #pragma unroll
        for (int t = 0; t < 4; ++t) {
            acc[0][t] = __builtin_amdgcn_wmma_f32_16x16x32_bf16(
                false, af0, false, bfr[t], (short)0, acc[0][t], false, false);
            acc[1][t] = __builtin_amdgcn_wmma_f32_16x16x32_bf16(
                false, af1, false, bfr[t], (short)0, acc[1][t], false, false);
        }

        wait_async0();
        __syncthreads();
    }

    // ---- epilogue ----
    const int nlocal = lane & 15;
    const int mhalf  = (lane >> 4) * 8;
    #pragma unroll
    for (int g = 0; g < 2; ++g) {
        const int mbase = block_m + wm * 32 + g * 16 + mhalf;
        float sn[8];
        if (agg) {
            #pragma unroll
            for (int r = 0; r < 8; ++r) {
                float dv = dinv[min(mbase + r, M - 1)];
                sn[r] = dv * dv;
            }
        }
        #pragma unroll
        for (int t = 0; t < 4; ++t) {
            int gn = block_n + wn * 64 + t * 16 + nlocal;
            float bv = bias ? bias[gn] : 0.0f;
            float cv = (agg != nullptr) ? cb[gn] : 0.0f;
            #pragma unroll
            for (int r = 0; r < 8; ++r) {
                int gm = mbase + r;
                if (gm < M) {
                    size_t idx = (size_t)gm * Nn + gn;
                    float c = acc[g][t][r] + bv;
                    C[idx] = c;
                    if (agg) agg[idx] = c * sn[r] + cv;
                }
            }
        }
    }
}

// ============================================================================
// GENERIC GEMM (input projection only, K=78): f32 A/B staged to bf16 in LDS,
// branchless clamp+select padding. 64x128 tile. Dual-writes f32 C and bf16 Cb.
// ============================================================================
#define PBM 64
__global__ __launch_bounds__(256)
void k_gemm_proj(const float* __restrict__ A, const float* __restrict__ B,
                 const float* __restrict__ bias, float* __restrict__ C,
                 __bf16* __restrict__ Cb, int M, int K, int Nn) {
    __shared__ __bf16 sA[PBM * LDA];
    __shared__ __bf16 sB[FBK * LDB];

    const int tid  = threadIdx.x;
    const int lane = tid & 31;
    const int wave = tid >> 5;
    const int wm   = wave & 3;
    const int wn   = wave >> 2;
    const int block_m = blockIdx.y * PBM;
    const int block_n = blockIdx.x * FBN;

    v8f zero = {0.f,0.f,0.f,0.f,0.f,0.f,0.f,0.f};
    v8f acc[4]; acc[0]=zero; acc[1]=zero; acc[2]=zero; acc[3]=zero;

    const int ksteps = (K + FBK - 1) / FBK;
    for (int ks = 0; ks < ksteps; ++ks) {
        const int k0 = ks * FBK;
        #pragma unroll
        for (int i = tid; i < PBM * FBK; i += 256) {
            int r = i >> 5, c = i & 31;
            int gk = k0 + c;
            float v = A[(size_t)min(block_m + r, M - 1) * K + min(gk, K - 1)];
            v = (gk < K) ? v : 0.0f;
            sA[r * LDA + c] = (__bf16)v;
        }
        #pragma unroll
        for (int i = tid; i < FBK * FBN; i += 256) {
            int r = i >> 7, c = i & 127;
            int gk = k0 + r;
            float v = B[(size_t)min(gk, K - 1) * Nn + block_n + c];
            v = (gk < K) ? v : 0.0f;
            sB[r * LDB + c] = (__bf16)v;
        }
        __syncthreads();

        const int half = lane >> 4;
        const int am   = wm * 16 + (lane & 15);
        v16bf afrag;
        #pragma unroll
        for (int j = 0; j < 8; ++j) {
            afrag[j]     = sA[am * LDA + half * 8 + j];
            afrag[8 + j] = sA[am * LDA + 16 + half * 8 + j];
        }
        #pragma unroll
        for (int t = 0; t < 4; ++t) {
            const int nb = wn * 64 + t * 16;
            v16bf bfrag;
            #pragma unroll
            for (int j = 0; j < 8; ++j) {
                bfrag[j]     = sB[(half * 8 + j) * LDB + nb + (lane & 15)];
                bfrag[8 + j] = sB[(16 + half * 8 + j) * LDB + nb + (lane & 15)];
            }
            acc[t] = __builtin_amdgcn_wmma_f32_16x16x32_bf16(
                false, afrag, false, bfrag, (short)0, acc[t], false, false);
        }
        __syncthreads();
    }

    const int nlocal = lane & 15;
    const int mhalf  = (lane >> 4) * 8;
    #pragma unroll
    for (int t = 0; t < 4; ++t) {
        int gn = block_n + wn * 64 + t * 16 + nlocal;
        float bv = bias[gn];
        #pragma unroll
        for (int r = 0; r < 8; ++r) {
            int gm = block_m + wm * 16 + mhalf + r;
            if (gm < M) {
                size_t idx = (size_t)gm * Nn + gn;
                float c = acc[t][r] + bv;
                C[idx]  = c;
                Cb[idx] = (__bf16)c;
            }
        }
    }
}

// ---------------- small kernels ----------------
__global__ void k_f2b(const float* __restrict__ in, __bf16* __restrict__ out, size_t n) {
    size_t i = (size_t)blockIdx.x * blockDim.x + threadIdx.x;
    if (i < n) out[i] = (__bf16)in[i];
}

__global__ void k_degree(const int* __restrict__ dst, float* __restrict__ deg, int E) {
    int e = blockIdx.x * blockDim.x + threadIdx.x;
    if (e < E) atomicAdd(&deg[dst[e]], 1.0f);
}

__global__ void k_dinv(const float* __restrict__ deg, float* __restrict__ dinv, int n) {
    int i = blockIdx.x * blockDim.x + threadIdx.x;
    if (i < n) dinv[i] = rsqrtf(deg[i] + 1.0f);
}

// per-edge scatter: agg[dst] += xw[src] * (dinv[src]*dinv[dst]); one block per edge
__global__ void k_scatter(const float* __restrict__ xw, const int* __restrict__ src,
                          const int* __restrict__ dst, const float* __restrict__ dinv,
                          float* __restrict__ agg, int E, int Dd) {
    int e = blockIdx.x;
    int s = src[e], d = dst[e];
    float c = dinv[s] * dinv[d];
    const float* xs = xw  + (size_t)s * Dd;
    float*       ad = agg + (size_t)d * Dd;
    for (int f = threadIdx.x; f < Dd; f += blockDim.x)
        atomicAdd(&ad[f], xs[f] * c);
}

// BN stats: each thread owns 3 fixed columns (768 = 3*256)
__global__ __launch_bounds__(256)
void k_bnstats(const float* __restrict__ agg, float* __restrict__ sum,
               float* __restrict__ sumsq, int N, int Dd) {
    int r0 = blockIdx.x * 256;
    int r1 = min(r0 + 256, N);
    float s[3]  = {0.f, 0.f, 0.f};
    float s2[3] = {0.f, 0.f, 0.f};
    for (int r = r0; r < r1; ++r) {
        const float* row = agg + (size_t)r * Dd;
        #pragma unroll
        for (int j = 0; j < 3; ++j) {
            float v = row[threadIdx.x + j * 256];
            s[j] += v; s2[j] += v * v;
        }
    }
    #pragma unroll
    for (int j = 0; j < 3; ++j) {
        int f = threadIdx.x + j * 256;
        atomicAdd(&sum[f], s[j]);
        atomicAdd(&sumsq[f], s2[j]);
    }
}

__global__ void k_bnfinal(const float* __restrict__ sum, const float* __restrict__ sumsq,
                          const float* __restrict__ g, const float* __restrict__ b,
                          float* __restrict__ colA, float* __restrict__ colB,
                          int N, int Dd) {
    int f = blockIdx.x * blockDim.x + threadIdx.x;
    if (f >= Dd) return;
    float inv_n = 1.0f / (float)N;
    float mu  = sum[f] * inv_n;
    float var = sumsq[f] * inv_n - mu * mu;
    float rstd = rsqrtf(var + BN_EPS);
    float a = rstd * g[f];
    colA[f] = a;
    colB[f] = b[f] - mu * a;
}

// h += relu(agg*colA + colB); also refresh bf16 copy of h
__global__ void k_update(const float* __restrict__ agg, const float* __restrict__ colA,
                         const float* __restrict__ colB, float* __restrict__ h,
                         __bf16* __restrict__ hb, int N, int Dd) {
    size_t i = (size_t)blockIdx.x * blockDim.x + threadIdx.x;
    if (i >= (size_t)N * Dd) return;
    int col = (int)(i % Dd);
    float hn = agg[i] * colA[col] + colB[col];
    float hv = h[i] + fmaxf(hn, 0.0f);
    h[i]  = hv;
    hb[i] = (__bf16)hv;
}

// ---------------- driver ----------------
extern "C" void kernel_launch(void* const* d_in, const int* in_sizes, int n_in,
                              void* d_out, int out_size, void* d_ws, size_t ws_size,
                              hipStream_t stream) {
    const float* x      = (const float*)d_in[0];
    const int*   eidx   = (const int*)  d_in[1];
    const float* in_w   = (const float*)d_in[2];
    const float* in_b   = (const float*)d_in[3];
    const float* conv_w = (const float*)d_in[4];
    const float* conv_b = (const float*)d_in[5];
    const float* bn_g   = (const float*)d_in[6];
    const float* bn_b   = (const float*)d_in[7];
    const float* out_w  = (const float*)d_in[8];
    const float* out_b  = (const float*)d_in[9];

    const int N = in_sizes[0] / D_IN;
    const int E = in_sizes[1] / 2;
    const int* src = eidx;
    const int* dst = eidx + E;

    // workspace carve-up: f32 arrays first, then bf16
    float* h    = (float*)d_ws;                    // N*DD
    float* agg  = h    + (size_t)N * DD;           // N*DD
    float* deg  = agg  + (size_t)N * DD;           // N
    float* dinv = deg  + N;                        // N
    float* sums = dinv + N;                        // 2*DD
    float* colA = sums + 2 * DD;                   // DD
    float* colB = colA + DD;                       // DD
    __bf16* hb  = (__bf16*)(colB + DD);            // N*DD
    __bf16* cwb = hb  + (size_t)N * DD;            // LL*DD*DD
    __bf16* owb = cwb + (size_t)LL * DD * DD;      // DD*DD
    float*  xw  = (float*)d_out;                   // scratch; overwritten by final GEMM

    // pre-convert weights to bf16 (once per launch)
    {
        size_t ncw = (size_t)LL * DD * DD;
        k_f2b<<<(int)((ncw + 255) / 256), 256, 0, stream>>>(conv_w, cwb, ncw);
        size_t now = (size_t)DD * DD;
        k_f2b<<<(int)((now + 255) / 256), 256, 0, stream>>>(out_w, owb, now);
    }

    // degree + normalization
    hipMemsetAsync(deg, 0, (size_t)N * sizeof(float), stream);
    k_degree<<<(E + 255) / 256, 256, 0, stream>>>(dst, deg, E);
    k_dinv  <<<(N + 255) / 256, 256, 0, stream>>>(deg, dinv, N);

    const dim3 gg_fast(DD / FBN, (N + FBM - 1) / FBM);
    const dim3 gg_proj(DD / FBN, (N + PBM - 1) / PBM);
    const int nd_blocks = (int)(((size_t)N * DD + 255) / 256);

    // input projection: h (f32) + hb (bf16) = x @ in_w + in_b
    k_gemm_proj<<<gg_proj, 256, 0, stream>>>(x, in_w, in_b, h, hb, N, D_IN, DD);

    for (int l = 0; l < LL; ++l) {
        // xw = hb @ conv_w_bf16[l]; fused agg = xw*self_norm + conv_b[l]
        k_gemm_fast<<<gg_fast, 256, 0, stream>>>(hb, cwb + (size_t)l * DD * DD,
                                                 nullptr, xw, agg, dinv,
                                                 conv_b + (size_t)l * DD, N, DD, DD);
        k_scatter<<<E, 256, 0, stream>>>(xw, src, dst, dinv, agg, E, DD);
        hipMemsetAsync(sums, 0, 2 * DD * sizeof(float), stream);
        k_bnstats<<<(N + 255) / 256, 256, 0, stream>>>(agg, sums, sums + DD, N, DD);
        k_bnfinal<<<(DD + 255) / 256, 256, 0, stream>>>(sums, sums + DD,
                                                        bn_g + (size_t)l * DD,
                                                        bn_b + (size_t)l * DD,
                                                        colA, colB, N, DD);
        k_update<<<nd_blocks, 256, 0, stream>>>(agg, colA, colB, h, hb, N, DD);
    }

    // output projection: d_out = hb @ out_w_bf16 + out_b
    k_gemm_fast<<<gg_fast, 256, 0, stream>>>(hb, owb, out_b, (float*)d_out,
                                             nullptr, nullptr, nullptr, N, DD, DD);
}
